// MyModel_13237089206734
// MI455X (gfx1250) — matrix operand order repaired and checked by reading, not compiled, not run on hardware
//
#include <hip/hip_runtime.h>
#include <hip/hip_bf16.h>

typedef __attribute__((ext_vector_type(16))) _Float16 v16h;
typedef __attribute__((ext_vector_type(8)))  _Float16 v8h;
typedef __attribute__((ext_vector_type(8)))  float    v8f;

#define HDIM  10
#define BSZ   4096
#define TSEQ  2048
#define FUTN  64
#define TFTOT (TSEQ + FUTN)   // 2112
#define CHUNK 64              // x / out staging chunk (TSEQ % CHUNK == 0, FUTN == CHUNK)

// Fast activations: hardware v_exp_f32 + v_rcp_f32 (approx, ~1ULP — far below the
// f16 quantization already present in the WMMA datapath). Both are overflow-safe.
__device__ __forceinline__ float fsig(float x) {
  return __builtin_amdgcn_rcpf(1.f + __expf(-x));          // exp(-x)->inf => rcp->0
}
__device__ __forceinline__ float ftanh(float x) {
  // tanh(x) = 1 - 2/(1 + e^{2x});  e^{2x}->inf => 1;  e^{2x}->0 => -1
  return 1.f - 2.f * __builtin_amdgcn_rcpf(1.f + __expf(2.f * x));
}

// One wave (32 lanes) per block; each block owns 16 batch rows for the full sequence.
__global__ __launch_bounds__(32) void lstm2_wmma_kernel(
    const float* __restrict__ X,
    const float* __restrict__ Wih1, const float* __restrict__ Whh1,
    const float* __restrict__ bih1, const float* __restrict__ bhh1,
    const float* __restrict__ Wih2, const float* __restrict__ Whh2,
    const float* __restrict__ bih2, const float* __restrict__ bhh2,
    const float* __restrict__ Wl,   const float* __restrict__ bl,
    float* __restrict__ Out)
{
  // wave-private LDS (one wave per workgroup)
  __shared__ _Float16 hbuf1[16][16];     // h1 state, row m, col j (cols 10..15 zero)
  __shared__ _Float16 hbuf2[16][16];     // h2 state
  __shared__ float    xtile[16][CHUNK];  // staged x chunk
  __shared__ float    otile[16][CHUNK];  // staged out chunk
  __shared__ float    xfeed[16];         // autoregressive feedback

  const int  lane = threadIdx.x;         // 0..31
  const int  nloc = lane & 15;           // N slot / hidden index j
  const bool hiH  = lane >= 16;          // lane half
  const int  m0   = blockIdx.x * 16;     // first batch row of this tile

  // ---------------------------------------------------------------
  // Pack weight matrices into WMMA B-operand layout, once.
  // B (f16, 32x16): lanes 0-15: elems 0..7 <-> K=0..7, elems 8..15 <-> K=16..23,
  //                 lanes 16-31: elems 0..7 <-> K=8..15, elems 8..15 <-> K=24..31.
  // Layer1 B rows: K=0..9 -> Whh1^T, K=10 -> Wih1, K=11 -> bih1+bhh1.
  // Layer2 B rows: K=0..9 -> Wih2^T, K=10..19 -> Whh2^T, K=20 -> bih2+bhh2.
  // N tiles g=0..3 are gate types i,f,g,o (gate row n = g*10 + nloc).
  // ---------------------------------------------------------------
  v16h B1[4], B2[4];
  #pragma unroll
  for (int g = 0; g < 4; ++g) {
    v16h b1v = {}; v16h b2v = {};
    if (nloc < HDIM) {
      const int n = g * HDIM + nloc;
      const float bb1 = bih1[n] + bhh1[n];
      const float bb2 = bih2[n] + bhh2[n];
      #pragma unroll
      for (int e = 0; e < 16; ++e) {
        const int k = (e < 8 ? 0 : 8) + (hiH ? 8 : 0) + e;
        float w1 = 0.f, w2 = 0.f;
        if (k < HDIM)             w1 = Whh1[n * HDIM + k];
        else if (k == HDIM)       w1 = Wih1[n];           // input column (IN==1)
        else if (k == HDIM + 1)   w1 = bb1;               // bias slot
        if (k < HDIM)             w2 = Wih2[n * HDIM + k];
        else if (k < 2 * HDIM)    w2 = Whh2[n * HDIM + (k - HDIM)];
        else if (k == 2 * HDIM)   w2 = bb2;               // bias slot
        b1v[e] = (_Float16)w1;
        b2v[e] = (_Float16)w2;
      }
    }
    B1[g] = b1v; B2[g] = b2v;
  }
  const float wlv = (nloc < HDIM) ? Wl[nloc] : 0.f;
  const float blv = bl[0];

  // zero-init LDS state (h1 = h2 = 0)
  if (lane < 16) {
    #pragma unroll
    for (int j = 0; j < 16; ++j) { hbuf1[lane][j] = (_Float16)0.f; hbuf2[lane][j] = (_Float16)0.f; }
    xfeed[lane] = 0.f;
  }
  __syncthreads();

  // cell state in f32 VGPRs, C-layout: c[r] <-> (m = r + 8*hiH, j = nloc)
  float c1[8] = {0.f,0.f,0.f,0.f,0.f,0.f,0.f,0.f};
  float c2[8] = {0.f,0.f,0.f,0.f,0.f,0.f,0.f,0.f};

  const v8f      zacc = {};
  const _Float16 zh   = (_Float16)0.f;
  const _Float16 oneh = (_Float16)1.f;
  const int      mrow = nloc;            // batch row served by this lane's A slots
  // per-lane base element offset into the h row for the A-operand b128 load:
  // lanes 0-15 read elems 0..7 (K=0..7), lanes 16-31 read elems 8..15 (K=8..15)
  const int abase = hiH ? 8 : 0;

  #pragma clang loop unroll(disable)
  for (int t = 0; t < TFTOT; ++t) {
    const int tc = t & (CHUNK - 1);

    // stage next x chunk (main phase only), coalesced along t
    if (t < TSEQ && tc == 0) {
      for (int idx = lane; idx < 16 * CHUNK; idx += 32) {
        const int mm = idx >> 6, cc = idx & (CHUNK - 1);
        xtile[mm][cc] = X[(size_t)(m0 + mm) * TSEQ + t + cc];
      }
      __syncthreads();
    }

    // current input value (main phase: staged x; future phase: fed-back output)
    const float* xsrc = (t < TSEQ) ? &xtile[mrow][tc] : &xfeed[mrow];
    const _Float16 xh = (_Float16)(*xsrc);

    // ---------------- Layer 1 (branchless A build) ----------------
    // A (f16 16x32): lanes 0-15: elems 0..7 <-> K=0..7, elems 8..15 <-> K=16..23
    //                lanes 16-31: elems 0..7 <-> K=8..15, elems 8..15 <-> K=24..31
    const v8h ld1 = *(const v8h*)&hbuf1[mrow][abase];   // aligned ds_load_b128
    v16h a1 = {};
    a1[0] = ld1[0];
    a1[1] = ld1[1];
    a1[2] = hiH ? xh   : ld1[2];   // hi: K=10 (input)
    a1[3] = hiH ? oneh : ld1[3];   // hi: K=11 (bias slot)
    a1[4] = hiH ? zh : ld1[4];
    a1[5] = hiH ? zh : ld1[5];
    a1[6] = hiH ? zh : ld1[6];
    a1[7] = hiH ? zh : ld1[7];
    // elems 8..15 (K=16..23 / 24..31) stay 0

    v8f gI = __builtin_amdgcn_wmma_f32_16x16x32_f16(false, a1, false, B1[0], (short)0, zacc, false, false);
    v8f gF = __builtin_amdgcn_wmma_f32_16x16x32_f16(false, a1, false, B1[1], (short)0, zacc, false, false);
    v8f gG = __builtin_amdgcn_wmma_f32_16x16x32_f16(false, a1, false, B1[2], (short)0, zacc, false, false);
    v8f gO = __builtin_amdgcn_wmma_f32_16x16x32_f16(false, a1, false, B1[3], (short)0, zacc, false, false);

    float h1n[8];
    #pragma unroll
    for (int r = 0; r < 8; ++r) {
      const float cn = fsig(gF[r]) * c1[r] + fsig(gI[r]) * ftanh(gG[r]);
      c1[r]  = cn;
      h1n[r] = fsig(gO[r]) * ftanh(cn);
    }
    if (nloc < HDIM) {
      #pragma unroll
      for (int r = 0; r < 8; ++r) hbuf1[r + (hiH ? 8 : 0)][nloc] = (_Float16)h1n[r];
    }
    __syncthreads();   // h1(new) visible before layer-2 A build

    // ---------------- Layer 2 (branchless A build) ----------------
    // K layout: 0..9 = h1 (new), 10..19 = h2 (prev), 20 = 1.0 (bias)
    const v8h l1  = *(const v8h*)&hbuf1[mrow][abase];   // h1 row (this lane's K half)
    const v8h l2a = *(const v8h*)&hbuf2[mrow][0];       // h2[0..7]
    const v8h l2b = *(const v8h*)&hbuf2[mrow][8];       // h2[8..15] (10..15 zero)
    v16h a2 = {};
    a2[0]  = l1[0];                    // lo: h1[0..1] | hi: h1[8..9]
    a2[1]  = l1[1];
    a2[2]  = hiH ? l2a[0] : l1[2];     // hi: K=10..15 -> h2[0..5]
    a2[3]  = hiH ? l2a[1] : l1[3];
    a2[4]  = hiH ? l2a[2] : l1[4];
    a2[5]  = hiH ? l2a[3] : l1[5];
    a2[6]  = hiH ? l2a[4] : l1[6];
    a2[7]  = hiH ? l2a[5] : l1[7];
    a2[8]  = hiH ? zh : l2a[6];        // lo: K=16..19 -> h2[6..9]
    a2[9]  = hiH ? zh : l2a[7];
    a2[10] = hiH ? zh : l2b[0];
    a2[11] = hiH ? zh : l2b[1];
    a2[12] = hiH ? zh : oneh;          // lo: K=20 (bias slot)
    // elems 13..15 stay 0

    v8f hI = __builtin_amdgcn_wmma_f32_16x16x32_f16(false, a2, false, B2[0], (short)0, zacc, false, false);
    v8f hF = __builtin_amdgcn_wmma_f32_16x16x32_f16(false, a2, false, B2[1], (short)0, zacc, false, false);
    v8f hG = __builtin_amdgcn_wmma_f32_16x16x32_f16(false, a2, false, B2[2], (short)0, zacc, false, false);
    v8f hO = __builtin_amdgcn_wmma_f32_16x16x32_f16(false, a2, false, B2[3], (short)0, zacc, false, false);

    float h2n[8];
    #pragma unroll
    for (int r = 0; r < 8; ++r) {
      const float cn = fsig(hF[r]) * c2[r] + fsig(hI[r]) * ftanh(hG[r]);
      c2[r]  = cn;
      h2n[r] = fsig(hO[r]) * ftanh(cn);
    }
    if (nloc < HDIM) {
      #pragma unroll
      for (int r = 0; r < 8; ++r) hbuf2[r + (hiH ? 8 : 0)][nloc] = (_Float16)h2n[r];
    }

    // ---------------- output head: out[m] = sum_j Wl[j]*h2[m][j] + bl ----------------
    float p[8];
    #pragma unroll
    for (int r = 0; r < 8; ++r) p[r] = wlv * h2n[r];
    #pragma unroll
    for (int mask = 1; mask < 16; mask <<= 1) {
      #pragma unroll
      for (int r = 0; r < 8; ++r) p[r] += __shfl_xor(p[r], mask, 32);
    }
    if (nloc == 0) {                       // lanes 0 and 16 carry the reduced values
      #pragma unroll
      for (int r = 0; r < 8; ++r) {
        const int   mm = r + (hiH ? 8 : 0);
        const float ov = p[r] + blv;
        otile[mm][tc] = ov;
        xfeed[mm]     = ov;                // autoregressive feedback
      }
    }
    __syncthreads();   // h2/xfeed/otile visible before next iteration / flush

    // flush staged outputs, coalesced along t
    if (tc == CHUNK - 1) {
      for (int idx = lane; idx < 16 * CHUNK; idx += 32) {
        const int mm = idx >> 6, cc = idx & (CHUNK - 1);
        Out[(size_t)(m0 + mm) * TFTOT + (t - (CHUNK - 1)) + cc] = otile[mm][cc];
      }
      __syncthreads();
    }
  }
}

extern "C" void kernel_launch(void* const* d_in, const int* in_sizes, int n_in,
                              void* d_out, int out_size, void* d_ws, size_t ws_size,
                              hipStream_t stream) {
  (void)in_sizes; (void)n_in; (void)out_size; (void)d_ws; (void)ws_size;
  const float* X    = (const float*)d_in[0];
  const float* Wih1 = (const float*)d_in[1];
  const float* Whh1 = (const float*)d_in[2];
  const float* bih1 = (const float*)d_in[3];
  const float* bhh1 = (const float*)d_in[4];
  const float* Wih2 = (const float*)d_in[5];
  const float* Whh2 = (const float*)d_in[6];
  const float* bih2 = (const float*)d_in[7];
  const float* bhh2 = (const float*)d_in[8];
  const float* Wl   = (const float*)d_in[9];
  const float* bl   = (const float*)d_in[10];
  // d_in[11] = future (==64), baked in as FUTN

  dim3 grid(BSZ / 16);   // 256 independent batch tiles
  dim3 block(32);        // one wave per workgroup
  lstm2_wmma_kernel<<<grid, block, 0, stream>>>(
      X, Wih1, Whh1, bih1, bhh1, Wih2, Whh2, bih2, bhh2, Wl, bl, (float*)d_out);
}
